// BCNet_84954453115099
// MI455X (gfx1250) — matrix operand order repaired and checked by reading, not compiled, not run on hardware
//
#include <hip/hip_runtime.h>
#include <hip/hip_bf16.h>

typedef __attribute__((ext_vector_type(16))) __bf16 v16bf;
typedef __attribute__((ext_vector_type(2)))  __bf16 v2bf;
typedef __attribute__((ext_vector_type(8)))  float  v8f;

#define N_DIM 1024
#define D_DIM 128
#define C_DIM 1024
#define K_DIM 3072
#define H_OUT 8
#define B_SZ  16
#define KT    256

// LDS layout (dynamic shared):
//  vS  : 16 rows x 1032 bf16 (pitch 516 u32)   -> 33024 B  @ 0
//  vtS : 16 rows x 260 f32                     -> 16640 B  @ 33024
//  qbS : 128 rows x 264 bf16 (pitch 132 u32)   -> 67584 B  @ 49664
//  hmS : 8 x 256 f32                           ->  8192 B  @ 117248
#define VT_OFF 33024
#define QB_OFF 49664
#define HM_OFF 117248
#define SMEM_SZ 125440

union Frag16 {
  v16bf v;
  unsigned u[8];
  uint4 q4[2];
};

#if __has_builtin(__builtin_amdgcn_cvt_pk_bf16_f32)
// single v_cvt_pk_bf16_f32 per pair (not available on this toolchain per round-3 probe)
static __device__ __forceinline__ unsigned pk(float a, float b) {
  union { v2bf v; unsigned u; } c;
  c.v = __builtin_amdgcn_cvt_pk_bf16_f32(a, b);
  return c.u;
}
static __device__ __forceinline__ unsigned short f2bf(float f) {
  return (unsigned short)(pk(f, 0.f) & 0xFFFFu);
}
#else
// round-half-up + v_perm_b32 byte-permute: 3 independent VALU per pair
// out = {b[31:16], a[31:16]} -> bytes {b3,b2,a3,a2} -> selector 0x07060302 (src0=b, src1=a)
static __device__ __forceinline__ unsigned pk(float a, float b) {
  unsigned ua = __float_as_uint(a) + 0x8000u;
  unsigned ub = __float_as_uint(b) + 0x8000u;
  return __builtin_amdgcn_perm(ub, ua, 0x07060302u);
}
static __device__ __forceinline__ unsigned short f2bf(float f) {
  return (unsigned short)((__float_as_uint(f) + 0x8000u) >> 16);
}
#endif

static __device__ __forceinline__ v8f wmma_bf16(const Frag16& A, const Frag16& B, v8f C) {
  return __builtin_amdgcn_wmma_f32_16x16x32_bf16(false, A.v, false, B.v, (short)0, C, false, false);
}
static __device__ __forceinline__ void async_ld_b128(unsigned ldsaddr, unsigned long long gaddr) {
  // CDNA5: per-lane 16-byte copy from global memory into LDS, tracked with ASYNCcnt
  asm volatile("global_load_async_to_lds_b128 %0, %1, off"
               :: "v"(ldsaddr), "v"(gaddr) : "memory");
}
static __device__ __forceinline__ void wait_async0() {
#if __has_builtin(__builtin_amdgcn_s_wait_asynccnt)
  __builtin_amdgcn_s_wait_asynccnt(0);
#else
  asm volatile("s_wait_asynccnt 0x0" ::: "memory");
#endif
}

// ---------------- prep: weight-norm scalars ----------------
__global__ void zero2_kernel(float* s) { s[0] = 0.f; s[1] = 0.f; }

__global__ void sumsq_kernel(const float* __restrict__ W, int n, float* __restrict__ out) {
  __shared__ float red[256];
  float s = 0.f;
  for (size_t i = (size_t)blockIdx.x * 256 + threadIdx.x; i < (size_t)n; i += (size_t)gridDim.x * 256) {
    float x = W[i];
    s += x * x;
  }
  red[threadIdx.x] = s;
  __syncthreads();
  for (int off = 128; off > 0; off >>= 1) {
    if (threadIdx.x < off) red[threadIdx.x] += red[threadIdx.x + off];
    __syncthreads();
  }
  if (threadIdx.x == 0) atomicAdd(out, red[0]);
}

__global__ void finalize_kernel(float* s, const float* gv, const float* gq) {
  s[2] = gv[0] * rsqrtf(s[0]);
  s[3] = gq[0] * rsqrtf(s[1]);
}

// ---------------- prep: pack W (scaled, bf16) into B-fragment layout ----------------
// frag[ktile][cstep][lane][r] : lane covers col k = ktile*16 + (lane&15),
// VGPR r holds c = cstep*32 + (lane>>4)*16 + 2r, 2r+1   (contiguous order, same as the A operands)
__global__ void pack_w_kernel(const float* __restrict__ W, const float* __restrict__ scale_p,
                              unsigned* __restrict__ frag) {
  size_t tid = (size_t)blockIdx.x * 256 + threadIdx.x;   // 192*32*32 = 196608 threads
  float s = scale_p[0];
  int lane = (int)(tid & 31);
  int cs   = (int)((tid >> 5) & 31);
  int kt   = (int)(tid >> 10);
  int k    = kt * 16 + (lane & 15);
  int cbase = cs * 32 + (lane >> 4) * 16;
  const float4* wrow = (const float4*)(W + (size_t)k * C_DIM + cbase);
  unsigned* o = frag + tid * 8;
#pragma unroll
  for (int j = 0; j < 4; ++j) {
    float4 f = wrow[j];
    o[2 * j]     = pk(f.x * s, f.y * s);
    o[2 * j + 1] = pk(f.z * s, f.w * s);
  }
}

// ---------------- prep: q_ = q @ WqN^T + bq  (bf16 out, layout [b*128+d][K]) ----------------
__global__ void qproj_kernel(const float* __restrict__ q, const float* __restrict__ bq,
                             const unsigned* __restrict__ wq_frag,
                             unsigned short* __restrict__ qbf) {
  const int lane = threadIdx.x;
  const int m = lane & 15, hg = lane >> 4;
  const int k0 = blockIdx.x * 64;     // 48 blocks
  const int row0 = blockIdx.y * 16;   // 128 blocks (b*128+d rows)
  v8f acc[4];
#pragma unroll
  for (int i = 0; i < 4; ++i) acc[i] = (v8f)(0.f);

  const float4* qrow4 = (const float4*)(q + (size_t)(row0 + m) * C_DIM);
  // fragment base: uint4 units; ktile stride = 2048 uint4, cstep stride = 64 uint4
  const uint4* pB = (const uint4*)(wq_frag + (size_t)(k0 >> 4) * 8192) + lane * 2;

  Frag16 Bd[2][4];
#pragma unroll
  for (int kt = 0; kt < 4; ++kt) {
    Bd[0][kt].q4[0] = pB[kt * 2048];
    Bd[0][kt].q4[1] = pB[kt * 2048 + 1];
  }
#pragma unroll 4
  for (int cs = 0; cs < 32; ++cs) {
    const int cur = cs & 1, nxt = cur ^ 1;
    const uint4* pn = pB + ((cs + 1) & 31) * 64;   // prefetch next cstep
#pragma unroll
    for (int kt = 0; kt < 4; ++kt) {
      Bd[nxt][kt].q4[0] = pn[kt * 2048];
      Bd[nxt][kt].q4[1] = pn[kt * 2048 + 1];
    }
    Frag16 A;
    float4 f0 = qrow4[cs * 8 + hg * 4];
    float4 f1 = qrow4[cs * 8 + hg * 4 + 1];
    float4 f2 = qrow4[cs * 8 + hg * 4 + 2];
    float4 f3 = qrow4[cs * 8 + hg * 4 + 3];
    A.u[0] = pk(f0.x, f0.y); A.u[1] = pk(f0.z, f0.w);
    A.u[2] = pk(f1.x, f1.y); A.u[3] = pk(f1.z, f1.w);
    A.u[4] = pk(f2.x, f2.y); A.u[5] = pk(f2.z, f2.w);
    A.u[6] = pk(f3.x, f3.y); A.u[7] = pk(f3.z, f3.w);
#pragma unroll
    for (int kt = 0; kt < 4; ++kt)
      acc[kt] = wmma_bf16(A, Bd[cur][kt], acc[kt]);
  }
#pragma unroll
  for (int kt = 0; kt < 4; ++kt) {
    int k = k0 + kt * 16 + m;
    float bqv = bq[k];
#pragma unroll
    for (int r = 0; r < 8; ++r) {
      int row = row0 + hg * 8 + r;
      qbf[(size_t)row * K_DIM + k] = f2bf(acc[kt][r] + bqv);
    }
  }
}

// ---------------- fused: v_ tile GEMM + per-head logits GEMM ----------------
__global__ void fused_logits_kernel(const float* __restrict__ v,
                                    const float* __restrict__ bv,
                                    const unsigned* __restrict__ wv_frag,
                                    const unsigned short* __restrict__ qbf,
                                    const float* __restrict__ hm,
                                    const float* __restrict__ hbias,
                                    float* __restrict__ out) {
  extern __shared__ char smem[];
  unsigned* vS  = (unsigned*)smem;              // [16][516] u32 (bf16 pairs)
  float*    vtS = (float*)(smem + VT_OFF);      // [16][260] f32
  float*    hmS = (float*)(smem + HM_OFF);      // [8][256]  f32

  const int tid  = threadIdx.x;
  const int lane = tid & 31;
  const int w    = tid >> 5;       // wave == head
  const int b    = blockIdx.y;
  const int n0   = blockIdx.x * 16;
  const int m    = lane & 15;
  const int hg   = lane >> 4;

  // stage v rows -> LDS bf16 (read from HBM exactly once per row)
  const float4* vsrc = (const float4*)(v + ((size_t)b * N_DIM + n0) * C_DIM);
  for (int i = tid; i < 4096; i += 256) {
    int row = i >> 8;
    int c4  = i & 255;
    float4 f = vsrc[row * 256 + c4];
    unsigned* dst = vS + row * 516 + c4 * 2;
    dst[0] = pk(f.x, f.y);
    dst[1] = pk(f.z, f.w);
  }

  v8f acc[8];
#pragma unroll
  for (int i = 0; i < 8; ++i) acc[i] = (v8f)(0.f);

  const uint4* vrow4 = (const uint4*)(vS + m * 516);
  const uint4* qb4   = (const uint4*)(smem + QB_OFF);

  for (int k0 = 0; k0 < K_DIM; k0 += KT) {
    __syncthreads();   // protect vt/qb/hm from previous-iteration readers (and v staging, 1st iter)

    // async-stage q_ chunk [d][256] bf16 into LDS; copy overlaps the whole gemm1 phase
    {
      const unsigned short* qb0 = qbf + (size_t)b * D_DIM * K_DIM + k0;
      for (int i = tid; i < 4096; i += 256) {
        int d  = i >> 5;
        int w4 = i & 31;
        unsigned ldsa = (unsigned)(uintptr_t)(smem + QB_OFF + d * 528 + w4 * 16);
        unsigned long long ga = (unsigned long long)(uintptr_t)(qb0 + (size_t)d * K_DIM + w4 * 8);
        async_ld_b128(ldsa, ga);
      }
    }
    // stage this wave's hm chunk
    {
      const float* hrow = hm + (size_t)w * K_DIM + k0;
#pragma unroll
      for (int j = 0; j < 8; ++j)
        hmS[w * 256 + j * 32 + lane] = hrow[j * 32 + lane];
    }

    // ---- gemm1: vt[16][32 per wave] = v @ WvN^T, accumulate over c=1024 ----
    v8f c1a = (v8f)(0.f), c1b = (v8f)(0.f);
    const int ktg0 = (k0 >> 4) + 2 * w;   // this wave's two global k-tiles
    // fragment base: uint4 units; ktile stride = 2048 uint4, cstep stride = 64 uint4
    const uint4* pB = (const uint4*)(wv_frag + (size_t)ktg0 * 8192) + lane * 2;

    Frag16 B0[2], B1[2];
    B0[0].q4[0] = pB[0];    B0[0].q4[1] = pB[1];
    B1[0].q4[0] = pB[2048]; B1[0].q4[1] = pB[2049];
#pragma unroll 4
    for (int cs = 0; cs < 32; ++cs) {
      const int cur = cs & 1, nxt = cur ^ 1;
      const uint4* pn = pB + ((cs + 1) & 31) * 64;   // prefetch next cstep
      B0[nxt].q4[0] = pn[0];    B0[nxt].q4[1] = pn[1];
      B1[nxt].q4[0] = pn[2048]; B1[nxt].q4[1] = pn[2049];
      Frag16 A;                                      // from LDS: 2x ds_load_b128
      A.q4[0] = vrow4[cs * 4 + hg * 2];
      A.q4[1] = vrow4[cs * 4 + hg * 2 + 1];
      c1a = wmma_bf16(A, B0[cur], c1a);
      c1b = wmma_bf16(A, B1[cur], c1b);
    }
    // + bv, store vt tile to LDS (f32, unscaled by hm)
    {
      float bva = bv[ktg0 * 16 + m];
      float bvb = bv[(ktg0 + 1) * 16 + m];
      int ka = (2 * w) * 16 + m;
      int kb = (2 * w + 1) * 16 + m;
#pragma unroll
      for (int r = 0; r < 8; ++r) {
        int row = hg * 8 + r;
        vtS[row * 260 + ka] = c1a[r] + bva;
        vtS[row * 260 + kb] = c1b[r] + bvb;
      }
    }
    wait_async0();     // q_ chunk landed in LDS
    __syncthreads();

    // ---- gemm2: acc[h] += (vt * hm[h]) @ q_^T ----
    const float2* hrow2 = (const float2*)(hmS + w * 256);
    const float2* vt2   = (const float2*)(vtS + m * 260);
    for (int kc = 0; kc < 8; ++kc) {
      Frag16 A2;
      const int p0 = kc * 16 + hg * 4;     // float2 index of k-base
#pragma unroll
      for (int j = 0; j < 4; ++j) {
        float2 a0 = vt2[p0 + j];
        float2 h0 = hrow2[p0 + j];
        float2 a1 = vt2[p0 + 8 + j];
        float2 h1 = hrow2[p0 + 8 + j];
        A2.u[j]     = pk(a0.x * h0.x, a0.y * h0.y);
        A2.u[4 + j] = pk(a1.x * h1.x, a1.y * h1.y);
      }
      const int q0 = kc * 4 + hg;          // uint4 index of k-base within qb row
#pragma unroll
      for (int dt = 0; dt < 8; ++dt) {
        Frag16 B2;                          // from LDS: 2x ds_load_b128
        const uint4* qrow = qb4 + (dt * 16 + m) * 33;
        B2.q4[0] = qrow[q0];
        B2.q4[1] = qrow[q0 + 2];
        acc[dt] = wmma_bf16(A2, B2, acc[dt]);
      }
    }
  }

  // epilogue: + h_bias, coalesced stores (lanes 0..15 -> consecutive d)
  float hbv = hbias[w];
  float* obase = out + (((size_t)b * H_OUT + w) * N_DIM + n0) * D_DIM;
#pragma unroll
  for (int dt = 0; dt < 8; ++dt) {
    int d = dt * 16 + m;
#pragma unroll
    for (int r = 0; r < 8; ++r) {
      int n = hg * 8 + r;
      obase[(size_t)n * D_DIM + d] = acc[dt][r] + hbv;
    }
  }
}

extern "C" void kernel_launch(void* const* d_in, const int* in_sizes, int n_in,
                              void* d_out, int out_size, void* d_ws, size_t ws_size,
                              hipStream_t stream) {
  const float* v  = (const float*)d_in[0];
  const float* q  = (const float*)d_in[1];
  const float* Wv = (const float*)d_in[2];
  const float* gv = (const float*)d_in[3];
  const float* bv = (const float*)d_in[4];
  const float* Wq = (const float*)d_in[5];
  const float* gq = (const float*)d_in[6];
  const float* bq = (const float*)d_in[7];
  const float* hm = (const float*)d_in[8];
  const float* hb = (const float*)d_in[9];
  float* out = (float*)d_out;

  char* ws = (char*)d_ws;
  float* scales = (float*)ws;                                     // [ssv, ssq, sv, sq]
  const size_t FRAG_BYTES = (size_t)192 * 32 * 32 * 8 * 4;        // 6,291,456
  unsigned* wv_frag = (unsigned*)(ws + 256);
  unsigned* wq_frag = (unsigned*)(ws + 256 + FRAG_BYTES);
  unsigned short* qbf = (unsigned short*)(ws + 256 + 2 * FRAG_BYTES);  // 2048*3072 bf16

  zero2_kernel<<<1, 1, 0, stream>>>(scales);
  sumsq_kernel<<<512, 256, 0, stream>>>(Wv, K_DIM * C_DIM, scales + 0);
  sumsq_kernel<<<512, 256, 0, stream>>>(Wq, K_DIM * C_DIM, scales + 1);
  finalize_kernel<<<1, 1, 0, stream>>>(scales, gv, gq);

  pack_w_kernel<<<768, 256, 0, stream>>>(Wv, scales + 2, wv_frag);
  pack_w_kernel<<<768, 256, 0, stream>>>(Wq, scales + 3, wq_frag);

  qproj_kernel<<<dim3(48, 128), 32, 0, stream>>>(q, bq, wq_frag, qbf);

  (void)hipFuncSetAttribute((const void*)fused_logits_kernel,
                            hipFuncAttributeMaxDynamicSharedMemorySize, SMEM_SZ);
  fused_logits_kernel<<<dim3(64, 16), 256, SMEM_SZ, stream>>>(v, bv, wv_frag, qbf, hm, hb, out);
}
